// GAT_3186865734044
// MI455X (gfx1250) — matrix operand order repaired and checked by reading, not compiled, not run on hardware
//
#include <hip/hip_runtime.h>

#define NNODES 50000
#define NPAD   50048            // round up to multiple of 64 (GEMM block = 64 rows)
#define NEDGES 640000
#define ETOT   (NNODES + NEDGES)
#define HID    128
#define HEADS  8
#define CDIM   16
#define NEG_SLOPE 0.2f

typedef __attribute__((ext_vector_type(16))) __bf16 v16bf;
typedef __attribute__((ext_vector_type(8)))  __bf16 v8bf;
typedef __attribute__((ext_vector_type(8)))  float  v8f;

// ---------- helpers ----------
static __device__ __forceinline__ unsigned short f2bf(float f) {
  // round-to-nearest-even f32 -> bf16
  unsigned u = __float_as_uint(f);
  unsigned r = u + 0x7FFFu + ((u >> 16) & 1u);
  return (unsigned short)(r >> 16);
}
// monotonic float <-> uint encoding for atomic max on floats
static __device__ __forceinline__ unsigned fenc(float f) {
  unsigned u = __float_as_uint(f);
  return (u & 0x80000000u) ? ~u : (u | 0x80000000u);
}
static __device__ __forceinline__ float fdec(unsigned e) {
  unsigned u = (e & 0x80000000u) ? (e & 0x7FFFFFFFu) : ~e;
  return __uint_as_float(u);
}

// ---------- f32 -> bf16 conversion (row-major copy) ----------
__global__ void k_conv_bf16(const float* __restrict__ src,
                            unsigned short* __restrict__ dst, int total) {
  int t = blockIdx.x * blockDim.x + threadIdx.x;
  if (t < total) dst[t] = f2bf(src[t]);
}

// relu + convert (layer-1 output -> layer-2 bf16 input)
__global__ void k_relu_conv(const float* __restrict__ src,
                            unsigned short* __restrict__ dst, int total) {
  int t = blockIdx.x * blockDim.x + threadIdx.x;
  if (t < total) dst[t] = f2bf(fmaxf(src[t], 0.0f));
}

// zero-fill (for xb pad rows; keeps replays deterministic)
__global__ void k_zero_us(unsigned short* __restrict__ dst, int total) {
  int t = blockIdx.x * blockDim.x + threadIdx.x;
  if (t < total) dst[t] = 0;
}

// f32 [K x N] row-major -> bf16 [N x K] (N-major) so each lane's WMMA B fragment
// (16 consecutive K values for one column) is 32 contiguous bytes.
__global__ void k_conv_wT(const float* __restrict__ w,
                          unsigned short* __restrict__ wt) {
  int t = blockIdx.x * blockDim.x + threadIdx.x;   // t = k*128 + n
  if (t >= HID * HID) return;
  const int k = t >> 7, n = t & 127;
  wt[n * HID + k] = f2bf(w[t]);
}

// ---------- WMMA GEMM (dual): outl = xb@Wl, outr = xb@Wr ----------
// xb: bf16 row-major [NPAD x 128]; wtl/wtr: bf16 N-major [128(N) x 128(K)];
// outl/outr: f32 [NPAD x 128].  All buffers padded -> no guards anywhere.
// block = 256 thr = 8 waves; block handles 4 M-tiles (64 rows);
// wave w owns cols [16w, 16w+16) and keeps all 8 B fragments in VGPRs.
__global__ void __launch_bounds__(256)
k_gemm2_bf16_wmma(const unsigned short* __restrict__ xb,
                  const unsigned short* __restrict__ wtl,
                  const unsigned short* __restrict__ wtr,
                  float* __restrict__ outl, float* __restrict__ outr) {
  const int wv   = threadIdx.x >> 5;
  const int lane = threadIdx.x & 31;
  const int half = (lane < 16) ? 0 : 1;
  const int l16  = lane & 15;
  const int mbase = blockIdx.x * 64;
  const int ncol  = wv * 16 + l16;

  // Preload B fragments: 32x16 bf16 per K-chunk; lane holds column `ncol`,
  // lanes<16 K=kk*32+0..15, lanes>=16 K=kk*32+16..31 -> contiguous in N-major W.
  v16bf bl[4], br[4];
  {
    const unsigned short* pl = wtl + (size_t)ncol * HID + half * 16;
    const unsigned short* pr = wtr + (size_t)ncol * HID + half * 16;
#pragma unroll
    for (int kk = 0; kk < 4; ++kk) {
      bl[kk] = *(const v16bf*)(pl + kk * 32);
      br[kk] = *(const v16bf*)(pr + kk * 32);
    }
  }

#pragma unroll
  for (int m = 0; m < 4; ++m) {
    const int arow = mbase + m * 16 + l16;
    v8f accl = {}, accr = {};
#pragma unroll
    for (int kk = 0; kk < 4; ++kk) {
      // A fragment 16x32 bf16: per lane two contiguous 16B chunks:
      // K = kbase..kbase+7 and kbase+8..kbase+15, kbase = kk*32 + 16*half
      const unsigned short* ap = xb + (size_t)arow * HID + kk * 32 + half * 16;
      v8bf alo = *(const v8bf*)(ap);
      v8bf ahi = *(const v8bf*)(ap + 8);
      v16bf a = __builtin_shufflevector(alo, ahi,
                                        0, 1, 2, 3, 4, 5, 6, 7,
                                        8, 9, 10, 11, 12, 13, 14, 15);
      accl = __builtin_amdgcn_wmma_f32_16x16x32_bf16(
          false, a, false, bl[kk], (short)0, accl, false, false);
      accr = __builtin_amdgcn_wmma_f32_16x16x32_bf16(
          false, a, false, br[kk], (short)0, accr, false, false);
    }
    // C/D: VGPR r -> row rbase+r, col ncol; rows padded -> unguarded stores
    const size_t obase = (size_t)(mbase + m * 16 + half * 8) * HID + ncol;
    float* pol = outl + obase;
    float* por = outr + obase;
#pragma unroll
    for (int r = 0; r < 8; ++r) {
      pol[r * HID] = accl[r];
      por[r * HID] = accr[r];
    }
  }
}

// ---------- per-layer init: out = bias, m = -inf(enc 0), denom = 0 ----------
__global__ void k_init(float* __restrict__ out, const float* __restrict__ bias,
                       unsigned* __restrict__ menc, float* __restrict__ denom) {
  const int n = blockIdx.x;
  const int j = threadIdx.x;
  out[(size_t)n * HID + j] = bias[j];
  if (j < HEADS) {
    menc[n * HEADS + j]  = 0u;     // encodes below every finite float
    denom[n * HEADS + j] = 0.0f;
  }
}

// ---------- pass 1: logits + segment max ----------
__global__ void k_logits(const int* __restrict__ ei,
                         const float* __restrict__ xl, const float* __restrict__ xr,
                         const float* __restrict__ att,
                         float* __restrict__ logits, unsigned* __restrict__ menc) {
  const int t = blockIdx.x * blockDim.x + threadIdx.x;
  if (t >= ETOT * HEADS) return;
  const int e = t >> 3, h = t & 7;
  const int s = (e < NEDGES) ? ei[e] : (e - NEDGES);
  const int d = (e < NEDGES) ? ei[NEDGES + e] : (e - NEDGES);
  const float4* pl = (const float4*)(xl + (size_t)s * HID + h * CDIM);
  const float4* pr = (const float4*)(xr + (size_t)d * HID + h * CDIM);
  const float4* pa = (const float4*)(att + h * CDIM);
  float acc = 0.0f;
#pragma unroll
  for (int q = 0; q < 4; ++q) {
    float4 a = pl[q], b = pr[q], w = pa[q];
    float v;
    v = a.x + b.x; v = (v > 0.f) ? v : NEG_SLOPE * v; acc += v * w.x;
    v = a.y + b.y; v = (v > 0.f) ? v : NEG_SLOPE * v; acc += v * w.y;
    v = a.z + b.z; v = (v > 0.f) ? v : NEG_SLOPE * v; acc += v * w.z;
    v = a.w + b.w; v = (v > 0.f) ? v : NEG_SLOPE * v; acc += v * w.w;
  }
  logits[t] = acc;
  atomicMax(&menc[d * HEADS + h], fenc(acc));   // native global_atomic_max_u32
}

// ---------- pass 2: alpha = exp(logit - m[dst]); denom += alpha ----------
__global__ void k_alpha(const int* __restrict__ ei,
                        const unsigned* __restrict__ menc,
                        float* __restrict__ alpha, float* __restrict__ denom) {
  const int t = blockIdx.x * blockDim.x + threadIdx.x;
  if (t >= ETOT * HEADS) return;
  const int e = t >> 3, h = t & 7;
  const int d = (e < NEDGES) ? ei[NEDGES + e] : (e - NEDGES);
  const float a = __expf(alpha[t] - fdec(menc[d * HEADS + h]));
  alpha[t] = a;
  unsafeAtomicAdd(&denom[d * HEADS + h], a);    // native global_atomic_add_f32
}

// ---------- pass 3: out[dst] += (alpha/denom) * xl[src] ----------
__global__ void k_aggregate(const int* __restrict__ ei,
                            const float* __restrict__ xl,
                            const float* __restrict__ alpha,
                            const float* __restrict__ denom,
                            float* __restrict__ out) {
  const int t = blockIdx.x * blockDim.x + threadIdx.x;
  if (t >= ETOT * HEADS) return;
  const int e = t >> 3, h = t & 7;
  const int s = (e < NEDGES) ? ei[e] : (e - NEDGES);
  const int d = (e < NEDGES) ? ei[NEDGES + e] : (e - NEDGES);
  const float w = alpha[t] / denom[d * HEADS + h];
  const float4* pl = (const float4*)(xl + (size_t)s * HID + h * CDIM);
  float* po = out + (size_t)d * HID + h * CDIM;
#pragma unroll
  for (int q = 0; q < 4; ++q) {
    float4 a = pl[q];
    unsafeAtomicAdd(po + q * 4 + 0, w * a.x);
    unsafeAtomicAdd(po + q * 4 + 1, w * a.y);
    unsafeAtomicAdd(po + q * 4 + 2, w * a.z);
    unsafeAtomicAdd(po + q * 4 + 3, w * a.w);
  }
}

// ---------- host ----------
extern "C" void kernel_launch(void* const* d_in, const int* in_sizes, int n_in,
                              void* d_out, int out_size, void* d_ws, size_t ws_size,
                              hipStream_t stream) {
  const float* x    = (const float*)d_in[0];
  const int*   ei   = (const int*)d_in[1];      // [2,E]: src then dst
  const float* Wl1  = (const float*)d_in[2];
  const float* Wr1  = (const float*)d_in[3];
  const float* att1 = (const float*)d_in[4];
  const float* b1   = (const float*)d_in[5];
  const float* Wl2  = (const float*)d_in[6];
  const float* Wr2  = (const float*)d_in[7];
  const float* att2 = (const float*)d_in[8];
  const float* b2   = (const float*)d_in[9];
  float* out = (float*)d_out;

  char* ws = (char*)d_ws;
  const size_t WELEM = (size_t)HID * HID;                    // 16384
  unsigned short* Wb0 = (unsigned short*)(ws);               // 4 x 32 KB bf16^T weights
  unsigned short* Wb1 = Wb0 + WELEM;
  unsigned short* Wb2 = Wb1 + WELEM;
  unsigned short* Wb3 = Wb2 + WELEM;
  size_t off = 4 * WELEM * sizeof(unsigned short);           // 131072
  unsigned short* xb = (unsigned short*)(ws + off);          // NPAD x 128 bf16
  off += (size_t)NPAD * HID * sizeof(unsigned short);
  float* xl = (float*)(ws + off);  off += (size_t)NPAD * HID * sizeof(float);
  float* xr = (float*)(ws + off);  off += (size_t)NPAD * HID * sizeof(float);
  float* h1 = (float*)(ws + off);  off += (size_t)NNODES * HID * sizeof(float);
  unsigned* menc = (unsigned*)(ws + off); off += (size_t)NNODES * HEADS * sizeof(unsigned);
  float* denom   = (float*)(ws + off);    off += (size_t)NNODES * HEADS * sizeof(float);
  float* alpha   = (float*)(ws + off);    // ETOT x 8, holds logits then alpha

  const int CB = 256;
  const int wBlocks  = (int)((WELEM + CB - 1) / CB);
  const int nhBlocks = (NNODES * HID + CB - 1) / CB;
  const int padElems = (NPAD - NNODES) * HID;                // 6144
  const int eBlocks  = (ETOT * HEADS + CB - 1) / CB;
  const dim3 gemmGrid(NPAD / 64);                            // 782, exact

  // weights -> transposed bf16 (tiny, L2-resident afterwards)
  k_conv_wT<<<wBlocks, CB, 0, stream>>>(Wl1, Wb0);
  k_conv_wT<<<wBlocks, CB, 0, stream>>>(Wr1, Wb1);
  k_conv_wT<<<wBlocks, CB, 0, stream>>>(Wl2, Wb2);
  k_conv_wT<<<wBlocks, CB, 0, stream>>>(Wr2, Wb3);
  // zero xb pad rows once per call (deterministic GEMM inputs for pad rows)
  k_zero_us<<<(padElems + CB - 1) / CB, CB, 0, stream>>>(
      xb + (size_t)NNODES * HID, padElems);

  // ===== layer 1 =====
  k_conv_bf16<<<nhBlocks, CB, 0, stream>>>(x, xb, NNODES * HID);
  k_gemm2_bf16_wmma<<<gemmGrid, 256, 0, stream>>>(xb, Wb0, Wb1, xl, xr);
  k_init<<<NNODES, HID, 0, stream>>>(h1, b1, menc, denom);
  k_logits<<<eBlocks, CB, 0, stream>>>(ei, xl, xr, att1, alpha, menc);
  k_alpha<<<eBlocks, CB, 0, stream>>>(ei, menc, alpha, denom);
  k_aggregate<<<eBlocks, CB, 0, stream>>>(ei, xl, alpha, denom, h1);

  // ===== layer 2 =====
  k_relu_conv<<<nhBlocks, CB, 0, stream>>>(h1, xb, NNODES * HID);
  k_gemm2_bf16_wmma<<<gemmGrid, 256, 0, stream>>>(xb, Wb2, Wb3, xl, xr);
  k_init<<<NNODES, HID, 0, stream>>>(out, b2, menc, denom);
  k_logits<<<eBlocks, CB, 0, stream>>>(ei, xl, xr, att2, alpha, menc);
  k_alpha<<<eBlocks, CB, 0, stream>>>(ei, menc, alpha, denom);
  k_aggregate<<<eBlocks, CB, 0, stream>>>(ei, xl, alpha, denom, out);
}
